// PatternConv3d_87617332838731
// MI455X (gfx1250) — matrix operand order repaired and checked by reading, not compiled, not run on hardware
//
#include <hip/hip_runtime.h>

// Problem constants (from reference):
//   x:      (B=2, C=64, T=16, H=112, W=112) f32
//   weight: (8, 8, Cout=64, C=64, 3, 3, 3)  f32
//   bias:   (8, 8, 64)                      f32
//   out:    (2, 64, 16, 112, 112)           f32
#define Bd    2
#define Cd    64
#define Td    16
#define Hd    112
#define Wd    112
#define COUT  64
#define HOd   14
#define WOd   14
#define NCOL  6272      // Bd*Td*HOd*WOd
#define NTILE 128       // columns per workgroup; 6272/128 = 49 exactly
#define NCHUNK 54       // 27 taps * 2 channel blocks of 32

typedef __attribute__((ext_vector_type(16))) __bf16 v16bf;
typedef __attribute__((ext_vector_type(8)))  float  v8f;

static __device__ inline unsigned int bfbits(__bf16 h) {
    union { __bf16 h; unsigned short u; } cv;
    cv.h = h;
    return (unsigned int)cv.u;
}

// Split f32 into bf16 hi (RNE) and bf16 lo = rne(f - (float)hi); pack pairs.
static __device__ inline void split2(float f0, float f1,
                                     unsigned int& hi, unsigned int& lo) {
    __bf16 h0 = (__bf16)f0, h1 = (__bf16)f1;
    __bf16 l0 = (__bf16)(f0 - (float)h0);
    __bf16 l1 = (__bf16)(f1 - (float)h1);
    hi = bfbits(h0) | (bfbits(h1) << 16);
    lo = bfbits(l0) | (bfbits(l1) << 16);
}

union FragU { uint4 q[2]; v16bf v; };

__global__ __launch_bounds__(256)
void pattern_conv3d_wmma(const float* __restrict__ x,
                         const float* __restrict__ weight,
                         const float* __restrict__ bias,
                         float* __restrict__ out)
{
    // Double-buffered LDS staging, laid out so each WMMA fragment is exactly
    // two ds_load_b128 per lane (contiguous k order: lane half h holds k = 16h..16h+15).
    //   A (weights): [co][k] bf16, k = c - c0 in [0,32)
    //   B (im2col) : [n ][k] bf16
    __shared__ alignas(16) unsigned short sAhi[2][COUT * 32];
    __shared__ alignas(16) unsigned short sAlo[2][COUT * 32];
    __shared__ alignas(16) unsigned short sBhi[2][NTILE * 32];
    __shared__ alignas(16) unsigned short sBlo[2][NTILE * 32];
    __shared__ float sBias[COUT];

    const int tid   = (int)threadIdx.x;
    const int phase = (int)blockIdx.y;       // 0..63
    const int ph    = phase >> 3;            // h phase
    const int pw    = phase & 7;             // w phase
    const int n0    = (int)blockIdx.x * NTILE;

    // ---- loader roles ----
    const int a_co  = tid >> 2;              // 0..63
    const int a_ci0 = (tid & 3) * 8;         // 0,8,16,24
    const int b_nj  = tid & 127;             // column within tile
    const int b_ci0 = (tid >> 7) * 16;       // 0 or 16

    int bn = n0 + b_nj;                      // decode loader column once
    const int l_wo = bn % WOd; bn /= WOd;
    const int l_ho = bn % HOd; bn /= HOd;
    const int l_t  = bn % Td;
    const int l_b  = bn / Td;

    // ---- compute role ----
    const int lane    = tid & 31;
    const int wave    = tid >> 5;            // 0..7 -> 16-column sub-tile
    const int halfsel = (lane >> 4) & 1;
    const int mlane   = lane & 15;
    int cn = n0 + wave * 16 + mlane;         // this lane's output column
    const int c_wo = cn % WOd; cn /= WOd;
    const int c_ho = cn % HOd; cn /= HOd;
    const int c_t  = cn % Td;
    const int c_b  = cn / Td;

    if (tid < COUT) sBias[tid] = bias[phase * COUT + tid];

    const int    sT = Hd * Wd;               // 12544
    const size_t sC = (size_t)Td * sT;       // 200704
    const size_t sB = (size_t)Cd * sC;

    const float* wa = weight + (size_t)phase * COUT * Cd * 27
                             + (size_t)a_co * (Cd * 27);
    const float* xbb = x + (size_t)l_b * sB;

    // Global fetch of one K-chunk into registers (software pipeline stage).
    auto fetchA = [&](int kc, float fa[8]) {
        const int tap = kc >> 1;
        const int c0  = (kc & 1) * 32;
        #pragma unroll
        for (int j = 0; j < 8; ++j)
            fa[j] = wa[(size_t)(c0 + a_ci0 + j) * 27 + tap];
    };
    auto fetchB = [&](int kc, float fb[16]) {
        const int tap = kc >> 1;
        const int c0  = (kc & 1) * 32;
        const int kd = tap / 9, kh = (tap / 3) % 3, kw = tap % 3;
        const int td = l_t + kd - 1;
        const int hr = l_ho * 8 + ph + kh - 7;   // max 106 < 112
        const int wr = l_wo * 8 + pw + kw - 7;
        if (((unsigned)td < (unsigned)Td) && (hr >= 0) && (wr >= 0)) {
            const float* xb = xbb + (size_t)td * sT + (size_t)hr * Wd + wr
                                  + (size_t)(c0 + b_ci0) * sC;
            #pragma unroll
            for (int j = 0; j < 16; ++j) fb[j] = xb[(size_t)j * sC];
        } else {
            #pragma unroll
            for (int j = 0; j < 16; ++j) fb[j] = 0.0f;
        }
    };

    v8f acc[4] = { v8f{}, v8f{}, v8f{}, v8f{} };

    float fa[8];
    float fb[16];
    fetchA(0, fa);
    fetchB(0, fb);

    for (int kc = 0; kc < NCHUNK; ++kc) {
        const int buf = kc & 1;

        // ---- convert + stage current chunk into LDS[buf] ----
        {
            unsigned int hi[4], lo[4];
            #pragma unroll
            for (int jp = 0; jp < 4; ++jp)
                split2(fa[2 * jp], fa[2 * jp + 1], hi[jp], lo[jp]);
            *((uint4*)(sAhi[buf] + a_co * 32 + a_ci0)) = make_uint4(hi[0], hi[1], hi[2], hi[3]);
            *((uint4*)(sAlo[buf] + a_co * 32 + a_ci0)) = make_uint4(lo[0], lo[1], lo[2], lo[3]);
        }
        {
            unsigned int hi[8], lo[8];
            #pragma unroll
            for (int jp = 0; jp < 8; ++jp)
                split2(fb[2 * jp], fb[2 * jp + 1], hi[jp], lo[jp]);
            unsigned short* dh = sBhi[buf] + b_nj * 32 + b_ci0;
            unsigned short* dl = sBlo[buf] + b_nj * 32 + b_ci0;
            *((uint4*)(dh))     = make_uint4(hi[0], hi[1], hi[2], hi[3]);
            *((uint4*)(dh + 8)) = make_uint4(hi[4], hi[5], hi[6], hi[7]);
            *((uint4*)(dl))     = make_uint4(lo[0], lo[1], lo[2], lo[3]);
            *((uint4*)(dl + 8)) = make_uint4(lo[4], lo[5], lo[6], lo[7]);
        }
        __syncthreads();   // single barrier per chunk (2-buffer pipeline)

        // ---- kick off next chunk's global loads; latency hides under WMMA ----
        if (kc + 1 < NCHUNK) {
            fetchA(kc + 1, fa);
            fetchB(kc + 1, fb);
        }

        // ---- gather ALL fragments first (20 independent ds_load_b128),
        //      then one dense 12-WMMA burst (4 independent 3-chains) ----
        FragU bh, bl, ah[4], al[4];
        {
            const uint4* pbh = (const uint4*)(sBhi[buf] + (wave * 16 + mlane) * 32);
            const uint4* pbl = (const uint4*)(sBlo[buf] + (wave * 16 + mlane) * 32);
            bh.q[0] = pbh[2 * halfsel]; bh.q[1] = pbh[2 * halfsel + 1];
            bl.q[0] = pbl[2 * halfsel]; bl.q[1] = pbl[2 * halfsel + 1];
        }
        #pragma unroll
        for (int mb = 0; mb < 4; ++mb) {
            const uint4* pah = (const uint4*)(sAhi[buf] + (mb * 16 + mlane) * 32);
            const uint4* pal = (const uint4*)(sAlo[buf] + (mb * 16 + mlane) * 32);
            ah[mb].q[0] = pah[2 * halfsel]; ah[mb].q[1] = pah[2 * halfsel + 1];
            al[mb].q[0] = pal[2 * halfsel]; al[mb].q[1] = pal[2 * halfsel + 1];
        }
        #pragma unroll
        for (int mb = 0; mb < 4; ++mb) {
            acc[mb] = __builtin_amdgcn_wmma_f32_16x16x32_bf16(
                false, ah[mb].v, false, bh.v, (short)0, acc[mb], false, false);
            acc[mb] = __builtin_amdgcn_wmma_f32_16x16x32_bf16(
                false, ah[mb].v, false, bl.v, (short)0, acc[mb], false, false);
            acc[mb] = __builtin_amdgcn_wmma_f32_16x16x32_bf16(
                false, al[mb].v, false, bh.v, (short)0, acc[mb], false, false);
        }
    }

    // ---- epilogue: C/D layout -> lane n = mlane, vgpr r -> M = r + 8*halfsel
    const int owh = (c_ho * 8 + ph) * Wd + (c_wo * 8 + pw);
    #pragma unroll
    for (int mb = 0; mb < 4; ++mb) {
        #pragma unroll
        for (int r = 0; r < 8; ++r) {
            const int co = mb * 16 + r + 8 * halfsel;
            const size_t oidx =
                (((size_t)c_b * COUT + co) * Td + c_t) * (size_t)sT + (size_t)owh;
            out[oidx] = acc[mb][r] + sBias[co];
        }
    }
}

extern "C" void kernel_launch(void* const* d_in, const int* in_sizes, int n_in,
                              void* d_out, int out_size, void* d_ws, size_t ws_size,
                              hipStream_t stream) {
    const float* x      = (const float*)d_in[0];
    const float* weight = (const float*)d_in[1];
    const float* bias   = (const float*)d_in[2];
    float* out          = (float*)d_out;
    (void)in_sizes; (void)n_in; (void)out_size; (void)d_ws; (void)ws_size;

    dim3 grid(NCOL / NTILE, 64, 1);   // 49 column tiles x 64 phases
    dim3 block(256, 1, 1);            // 8 wave32 waves
    pattern_conv3d_wmma<<<grid, block, 0, stream>>>(x, weight, bias, out);
}
